// GNN_10522669875448
// MI455X (gfx1250) — matrix operand order repaired and checked by reading, not compiled, not run on hardware
//
#include <hip/hip_runtime.h>
#include <hip/hip_bf16.h>

#define N_NODES  25000
#define N_EDGES  100000
#define N_GRAPHS 2048
#define F_NODE   133
#define F_EDGE   14
#define HIDDEN   512
#define DEPTH    4
#define K_EI     160   // 147 padded to mult of 32
#define K_Q      672   // 645 padded to mult of 32

typedef __attribute__((ext_vector_type(16))) __bf16 v16bf;
typedef __attribute__((ext_vector_type(8)))  float  v8f;

union FragAB { v16bf v; uint4 q[2]; };

__device__ __forceinline__ unsigned short f2bf(float f) {
  unsigned int u = __float_as_uint(f);
  u += 0x7FFFu + ((u >> 16) & 1u);           // round-to-nearest-even
  return (unsigned short)(u >> 16);
}
__device__ __forceinline__ float bf2f(unsigned short h) {
  return __uint_as_float(((unsigned int)h) << 16);
}

// ---------------------------------------------------------------------------
// Pack a [K,N] f32 weight matrix into WMMA-B frag-major bf16 layout, K padded.
// Element order: ((nt*numKt + kt)*32 + lane)*16 + j, where lane<16 holds
// K = kt*32 + j (j=0..15) of column nt*16+(lane&15); lanes 16..31 hold
// K = kt*32 + 16 + j. One contiguous 32B load per lane per B fragment.
// ---------------------------------------------------------------------------
__global__ void pack_b_kernel(const float* __restrict__ W,
                              unsigned short* __restrict__ out,
                              int K, int Kpad, int N) {
  const long long idx = (long long)blockIdx.x * blockDim.x + threadIdx.x;
  const int numKt = Kpad >> 5;
  const long long total = (long long)Kpad * N;
  if (idx >= total) return;
  const int j    = (int)(idx & 15);
  const int lane = (int)((idx >> 4) & 31);
  long long t    = idx >> 9;
  const int kt   = (int)(t % numKt);
  const int nt   = (int)(t / numKt);
  const int col  = nt * 16 + (lane & 15);
  const int k    = (kt << 5) + ((lane >> 4) << 4) + j;
  float v = (k < K && col < N) ? W[(long long)k * N + col] : 0.0f;
  out[idx] = f2bf(v);
}

// xin[e, 0..159] = [x[row[e]] (133) | edge_attr[e] (14) | zeros (13)] as bf16
__global__ void build_edge_input_kernel(const float* __restrict__ x,
                                        const float* __restrict__ ea,
                                        const int* __restrict__ row,
                                        unsigned short* __restrict__ xin) {
  const long long idx = (long long)blockIdx.x * blockDim.x + threadIdx.x;
  const long long total = (long long)N_EDGES * K_EI;
  if (idx >= total) return;
  const int c = (int)(idx % K_EI);
  const int e = (int)(idx / K_EI);
  float v = 0.0f;
  if (c < F_NODE)               v = x[(long long)row[e] * F_NODE + c];
  else if (c < F_NODE + F_EDGE) v = ea[(long long)e * F_EDGE + (c - F_NODE)];
  xin[idx] = f2bf(v);
}

__global__ void fill_zero_kernel(float* __restrict__ p, long long n) {
  const long long idx = (long long)blockIdx.x * blockDim.x + threadIdx.x;
  if (idx < n) p[idx] = 0.0f;
}

// a[col[e], :] += h[e, :]   (bf16 source, f32 atomic accumulate; a is L2-resident)
__global__ void seg_scatter_add_kernel(const unsigned short* __restrict__ h,
                                       const int* __restrict__ col,
                                       float* __restrict__ a) {
  const long long idx = (long long)blockIdx.x * blockDim.x + threadIdx.x;
  const long long total = (long long)N_EDGES * (HIDDEN / 8);
  if (idx >= total) return;
  const int ch = (int)(idx & 63);
  const int e  = (int)(idx >> 6);
  const unsigned short* hp = h + (long long)e * HIDDEN + ch * 8;
  ushort4 p0 = *(const ushort4*)(hp);
  ushort4 p1 = *(const ushort4*)(hp + 4);
  float* ap = a + (long long)col[e] * HIDDEN + ch * 8;
  atomicAdd(ap + 0, bf2f(p0.x)); atomicAdd(ap + 1, bf2f(p0.y));
  atomicAdd(ap + 2, bf2f(p0.z)); atomicAdd(ap + 3, bf2f(p0.w));
  atomicAdd(ap + 4, bf2f(p1.x)); atomicAdd(ap + 5, bf2f(p1.y));
  atomicAdd(ap + 6, bf2f(p1.z)); atomicAdd(ap + 7, bf2f(p1.w));
}

// m[e,:] = bf16( a[row[e],:] - h[e^1,:] )
__global__ void compute_msg_kernel(const float* __restrict__ a,
                                   const unsigned short* __restrict__ h,
                                   const int* __restrict__ row,
                                   unsigned short* __restrict__ m) {
  const long long idx = (long long)blockIdx.x * blockDim.x + threadIdx.x;
  const long long total = (long long)N_EDGES * (HIDDEN / 8);
  if (idx >= total) return;
  const int ch = (int)(idx & 63);
  const int e  = (int)(idx >> 6);
  const float* ap = a + (long long)row[e] * HIDDEN + ch * 8;
  const unsigned short* hr = h + (long long)(e ^ 1) * HIDDEN + ch * 8;
  unsigned short* mp = m + (long long)e * HIDDEN + ch * 8;
  float4 a0 = *(const float4*)(ap);
  float4 a1 = *(const float4*)(ap + 4);
  ushort4 r0 = *(const ushort4*)(hr);
  ushort4 r1 = *(const ushort4*)(hr + 4);
  ushort4 o0, o1;
  o0.x = f2bf(a0.x - bf2f(r0.x)); o0.y = f2bf(a0.y - bf2f(r0.y));
  o0.z = f2bf(a0.z - bf2f(r0.z)); o0.w = f2bf(a0.w - bf2f(r0.w));
  o1.x = f2bf(a1.x - bf2f(r1.x)); o1.y = f2bf(a1.y - bf2f(r1.y));
  o1.z = f2bf(a1.z - bf2f(r1.z)); o1.w = f2bf(a1.w - bf2f(r1.w));
  *(ushort4*)(mp)     = o0;
  *(ushort4*)(mp + 4) = o1;
}

// q[n, 0..671] = [x[n] (133) | s[n] (512) | zeros (27)] as bf16
__global__ void build_q_kernel(const float* __restrict__ x,
                               const float* __restrict__ s,
                               unsigned short* __restrict__ q) {
  const long long idx = (long long)blockIdx.x * blockDim.x + threadIdx.x;
  const long long total = (long long)N_NODES * K_Q;
  if (idx >= total) return;
  const int c = (int)(idx % K_Q);
  const int n = (int)(idx / K_Q);
  float v = 0.0f;
  if (c < F_NODE)                v = x[(long long)n * F_NODE + c];
  else if (c < F_NODE + HIDDEN)  v = s[(long long)n * HIDDEN + (c - F_NODE)];
  q[idx] = f2bf(v);
}

__global__ void cvt_f32_bf16_kernel(const float* __restrict__ in,
                                    unsigned short* __restrict__ out, long long n) {
  const long long idx = (long long)blockIdx.x * blockDim.x + threadIdx.x;
  if (idx < n) out[idx] = f2bf(in[idx]);
}

// ---------------------------------------------------------------------------
// Generic wave32 WMMA GEMM: C[M,512] = A_bf16[M,Kpad] @ Bpack_bf16 (+ epilogue)
//   mode 0: edge init  -> relu(acc+bias)              -> h0(bf16) and h(bf16)
//   mode 1: conv layer -> relu(acc+bias+h0)           -> h(bf16)
//   mode 2: e2n        -> relu(acc+bias)              -> atomicAdd pooled[batch[m]]
//   mode 3: ffn1       -> relu(acc+bias)              -> outF(f32)
// Block = 256 threads = 8 waves; block tile 128 (M) x 128 (N); wave tile 16x128.
// Double-buffered (ping-pong) A/B fragment loads; EXEC stays all-ones through
// the whole WMMA loop (out-of-range rows are clamped, masked only in epilogue).
// ---------------------------------------------------------------------------
__global__ __launch_bounds__(256)
void wmma_gemm_kernel(const unsigned short* __restrict__ A,
                      const unsigned short* __restrict__ Bp,
                      const float* __restrict__ bias,
                      int M, int Kpad, int mode,
                      unsigned short* __restrict__ outH,
                      unsigned short* __restrict__ outH0,
                      const unsigned short* __restrict__ h0in,
                      const int* __restrict__ batch,
                      float* __restrict__ pooled,
                      float* __restrict__ outF) {
  const int lane  = threadIdx.x & 31;
  const int wave  = threadIdx.x >> 5;
  const int half  = lane >> 4;        // 0: lanes 0-15, 1: lanes 16-31
  const int r     = lane & 15;
  const int mBase = blockIdx.x * 128 + wave * 16;
  if (mBase >= M) return;             // wave-uniform early exit (no barriers used)
  const int nBlock = blockIdx.y * 128;
  const int numKt  = Kpad >> 5;

  v8f acc[8];
#pragma unroll
  for (int i = 0; i < 8; ++i)
#pragma unroll
    for (int j = 0; j < 8; ++j) acc[i][j] = 0.0f;

  // Clamp instead of branch: a row of A only feeds the same row of C, and the
  // epilogue masks rows >= M, so clamped loads are safe and keep EXEC all-1s.
  const int rowA = (mBase + r < M) ? (mBase + r) : (M - 1);
  const unsigned short* Arow = A + (long long)rowA * Kpad;
  const int kA = (half << 4);                    // lanes 0-15: K 0..15; 16-31: K 16..31
  const unsigned short* bCol =
      Bp + ((long long)(blockIdx.y * 8) * numKt * 32) * 16 + (long long)lane * 16;
  const long long bNtStride = (long long)numKt * 512;   // elements between nt frags

  FragAB a0, a1, b0[8], b1[8];
  auto loadTile = [&](int kt, FragAB& af, FragAB (&bf)[8]) {
    const unsigned short* ap = Arow + (kt << 5) + kA;
    af.q[0] = *(const uint4*)(ap);
    af.q[1] = *(const uint4*)(ap + 8);
    const unsigned short* bp = bCol + (long long)kt * 512;  // kt stride = 32*16 elems
#pragma unroll
    for (int nt = 0; nt < 8; ++nt) {
      bf[nt].q[0] = *(const uint4*)(bp);
      bf[nt].q[1] = *(const uint4*)(bp + 8);
      bp += bNtStride;
    }
  };
  auto mmaTile = [&](const FragAB& af, const FragAB (&bf)[8]) {
#pragma unroll
    for (int nt = 0; nt < 8; ++nt)
      acc[nt] = __builtin_amdgcn_wmma_f32_16x16x32_bf16(
          false, af.v, false, bf[nt].v, (short)0, acc[nt], false, false);
  };

  loadTile(0, a0, b0);
  for (int kt = 0; kt < numKt; kt += 2) {
    __builtin_prefetch(Arow + ((kt + 2) << 5), 0, 0);   // global_prefetch_b8 on A stream
    if (kt + 1 < numKt) loadTile(kt + 1, a1, b1);
    mmaTile(a0, b0);
    if (kt + 1 < numKt) {
      if (kt + 2 < numKt) loadTile(kt + 2, a0, b0);
      mmaTile(a1, b1);
    }
  }

#pragma unroll
  for (int nt = 0; nt < 8; ++nt) {
    const int col = nBlock + nt * 16 + r;
    const float bv = bias[col];
#pragma unroll
    for (int v = 0; v < 8; ++v) {
      const int row = mBase + (half << 3) + v;   // C layout: VGPR v <-> M = v + 8*half
      if (row >= M) continue;
      const long long o = (long long)row * HIDDEN + col;
      float val = acc[nt][v] + bv;
      if (mode == 1) val += bf2f(h0in[o]);       // residual h0 before relu
      val = fmaxf(val, 0.0f);
      if (mode == 2) {
        atomicAdd(&pooled[(long long)batch[row] * HIDDEN + col], val);
      } else if (mode == 3) {
        outF[o] = val;
      } else {
        const unsigned short bvb = f2bf(val);
        outH[o] = bvb;
        if (mode == 0) outH0[o] = bvb;
      }
    }
  }
}

// out[g] = dot(t[g,:], W2[:,0]) + b2  (t already has ReLU applied)
__global__ void ffn2_kernel(const float* __restrict__ t,
                            const float* __restrict__ W2,
                            const float* __restrict__ b2,
                            float* __restrict__ out) {
  __shared__ float red[256];
  const int g = blockIdx.x, tid = threadIdx.x;
  float s = t[(long long)g * HIDDEN + tid] * W2[tid] +
            t[(long long)g * HIDDEN + tid + 256] * W2[tid + 256];
  red[tid] = s;
  __syncthreads();
  for (int off = 128; off > 0; off >>= 1) {
    if (tid < off) red[tid] += red[tid + off];
    __syncthreads();
  }
  if (tid == 0) out[g] = red[0] + b2[0];
}

// ---------------------------------------------------------------------------
extern "C" void kernel_launch(void* const* d_in, const int* in_sizes, int n_in,
                              void* d_out, int out_size, void* d_ws, size_t ws_size,
                              hipStream_t stream) {
  const float* x       = (const float*)d_in[0];
  const int*   ei      = (const int*)d_in[1];
  const float* eattr   = (const float*)d_in[2];
  const int*   batch   = (const int*)d_in[3];
  const float* W_ei    = (const float*)d_in[4];
  const float* b_ei    = (const float*)d_in[5];
  const float* W_conv  = (const float*)d_in[6];
  const float* b_conv  = (const float*)d_in[7];
  const float* W_e2n   = (const float*)d_in[8];
  const float* b_e2n   = (const float*)d_in[9];
  const float* W_ffn1  = (const float*)d_in[10];
  const float* b_ffn1  = (const float*)d_in[11];
  const float* W_ffn2  = (const float*)d_in[12];
  const float* b_ffn2  = (const float*)d_in[13];
  float* out = (float*)d_out;

  const int* row = ei;            // edge_index[0]
  const int* col = ei + N_EDGES;  // edge_index[1]

  // ---- workspace arena (256B aligned) ----
  size_t off = 0;
  char* base = (char*)d_ws;
  auto arena = [&](size_t bytes) -> char* {
    char* p = base + off;
    off += (bytes + 255) & ~(size_t)255;
    return p;
  };
  unsigned short* WbE = (unsigned short*)arena((size_t)K_EI * HIDDEN * 2);
  unsigned short* WbC = (unsigned short*)arena((size_t)DEPTH * HIDDEN * HIDDEN * 2);
  unsigned short* WbQ = (unsigned short*)arena((size_t)K_Q * HIDDEN * 2);
  unsigned short* WbF = (unsigned short*)arena((size_t)HIDDEN * HIDDEN * 2);
  unsigned short* xin = (unsigned short*)arena((size_t)N_EDGES * K_EI * 2);
  unsigned short* h0  = (unsigned short*)arena((size_t)N_EDGES * HIDDEN * 2);
  unsigned short* h   = (unsigned short*)arena((size_t)N_EDGES * HIDDEN * 2);
  unsigned short* msg = (unsigned short*)arena((size_t)N_EDGES * HIDDEN * 2);
  float*          agg = (float*)arena((size_t)N_NODES * HIDDEN * 4);
  unsigned short* q   = (unsigned short*)arena((size_t)N_NODES * K_Q * 2);
  float*          pooled  = (float*)arena((size_t)N_GRAPHS * HIDDEN * 4);
  unsigned short* pooledb = (unsigned short*)arena((size_t)N_GRAPHS * HIDDEN * 2);
  float*          tbuf    = (float*)arena((size_t)N_GRAPHS * HIDDEN * 4);
  (void)ws_size; (void)n_in; (void)in_sizes; (void)out_size;

  auto cdivll = [](long long a, long long b) -> int { return (int)((a + b - 1) / b); };
  const dim3 blk(256);

  // 1) pack all weights to WMMA-B frag-major bf16 (once per launch; tiny)
  pack_b_kernel<<<cdivll((long long)K_EI * HIDDEN, 256), blk, 0, stream>>>(W_ei, WbE, F_NODE + F_EDGE, K_EI, HIDDEN);
  for (int l = 0; l < DEPTH; ++l)
    pack_b_kernel<<<cdivll((long long)HIDDEN * HIDDEN, 256), blk, 0, stream>>>(
        W_conv + (long long)l * HIDDEN * HIDDEN, WbC + (long long)l * HIDDEN * HIDDEN, HIDDEN, HIDDEN, HIDDEN);
  pack_b_kernel<<<cdivll((long long)K_Q * HIDDEN, 256), blk, 0, stream>>>(W_e2n, WbQ, F_NODE + HIDDEN, K_Q, HIDDEN);
  pack_b_kernel<<<cdivll((long long)HIDDEN * HIDDEN, 256), blk, 0, stream>>>(W_ffn1, WbF, HIDDEN, HIDDEN, HIDDEN);

  // 2) edge-init input gather + GEMM -> h0, h
  build_edge_input_kernel<<<cdivll((long long)N_EDGES * K_EI, 256), blk, 0, stream>>>(x, eattr, row, xin);
  {
    dim3 grid(cdivll(N_EDGES, 128), HIDDEN / 128);
    wmma_gemm_kernel<<<grid, blk, 0, stream>>>(xin, WbE, b_ei, N_EDGES, K_EI, /*mode=*/0,
                                               h, h0, nullptr, nullptr, nullptr, nullptr);
  }

  // 3) DMPNN layers
  for (int l = 0; l < DEPTH; ++l) {
    fill_zero_kernel<<<cdivll((long long)N_NODES * HIDDEN, 256), blk, 0, stream>>>(agg, (long long)N_NODES * HIDDEN);
    seg_scatter_add_kernel<<<cdivll((long long)N_EDGES * 64, 256), blk, 0, stream>>>(h, col, agg);
    compute_msg_kernel<<<cdivll((long long)N_EDGES * 64, 256), blk, 0, stream>>>(agg, h, row, msg);
    dim3 grid(cdivll(N_EDGES, 128), HIDDEN / 128);
    wmma_gemm_kernel<<<grid, blk, 0, stream>>>(msg, WbC + (long long)l * HIDDEN * HIDDEN,
                                               b_conv + (long long)l * HIDDEN, N_EDGES, HIDDEN, /*mode=*/1,
                                               h, nullptr, h0, nullptr, nullptr, nullptr);
  }

  // 4) final node aggregation s, q = [x | s], e2n GEMM fused with graph pooling
  fill_zero_kernel<<<cdivll((long long)N_NODES * HIDDEN, 256), blk, 0, stream>>>(agg, (long long)N_NODES * HIDDEN);
  seg_scatter_add_kernel<<<cdivll((long long)N_EDGES * 64, 256), blk, 0, stream>>>(h, col, agg);
  build_q_kernel<<<cdivll((long long)N_NODES * K_Q, 256), blk, 0, stream>>>(x, agg, q);
  fill_zero_kernel<<<cdivll((long long)N_GRAPHS * HIDDEN, 256), blk, 0, stream>>>(pooled, (long long)N_GRAPHS * HIDDEN);
  {
    dim3 grid(cdivll(N_NODES, 128), HIDDEN / 128);
    wmma_gemm_kernel<<<grid, blk, 0, stream>>>(q, WbQ, b_e2n, N_NODES, K_Q, /*mode=*/2,
                                               nullptr, nullptr, nullptr, batch, pooled, nullptr);
  }

  // 5) FFN head
  cvt_f32_bf16_kernel<<<cdivll((long long)N_GRAPHS * HIDDEN, 256), blk, 0, stream>>>(
      pooled, pooledb, (long long)N_GRAPHS * HIDDEN);
  {
    dim3 grid(cdivll(N_GRAPHS, 128), HIDDEN / 128);
    wmma_gemm_kernel<<<grid, blk, 0, stream>>>(pooledb, WbF, b_ffn1, N_GRAPHS, HIDDEN, /*mode=*/3,
                                               nullptr, nullptr, nullptr, nullptr, nullptr, tbuf);
  }
  ffn2_kernel<<<N_GRAPHS, blk, 0, stream>>>(tbuf, W_ffn2, b_ffn2, out);
}
